// charRNN_53867479827171
// MI455X (gfx1250) — matrix operand order repaired and checked by reading, not compiled, not run on hardware
//
#include <hip/hip_runtime.h>

// CharRNN (GRU) for MI455X / gfx1250, wave32, bf16 WMMA (16x16x32, f32 accum).
//
// v3 changes (from v2 asm inspection: scratch spills + v_mov_b64 accumulator
// shuffling when using 32x64 tiles + array-rotated double buffers):
//  - Back to 16x64 tiles per wave (4 accumulators, 32 VGPRs) -- v1's clean
//    allocation -- but keep load/WMMA overlap via an explicit ping-pong:
//    the K-loop is manually unrolled 2x with two named fragment sets (even /
//    odd phase). No fragment rotation copies, no spills; loads for phase
//    k+32 issue before the phase-k WMMAs so s_wait_loadcnt covers loads from
//    half an iteration earlier instead of stalling cold.
//  - K is a multiple of 64 in all three GEMMs (256, 1024, 1024), so the
//    2x-unrolled loop needs no remainder handling beyond a peeled tail.
//
// Pipeline:
//   1) transpose+convert Wi,Wh,Wout -> bf16 [N,K] fragment-ready layout
//   2) embedding gather -> x bf16 [B*L, E]
//   3) WMMA GEMM: xw = x @ Wi   [16384,256]x[256,3072] -> bf16
//   4) 256 sequential gru_step launches; wave owns one 16x16 h-tile, computes
//      r/z/n gate tiles over K=1024 in registers, fuses gates. h kept fp32
//      for the recurrence; bf16 mirror feeds the next step's WMMA A operand.
//   5) WMMA GEMM: logits = y @ Wout + bout -> fp32 d_out

#define Bb   64
#define Ll   256
#define Vv   256
#define Ee   256
#define Hh   1024
#define G3H  (3 * Hh)     // 3072
#define BL   (Bb * Ll)    // 16384

typedef __attribute__((ext_vector_type(16))) __bf16        v16bf;
typedef __attribute__((ext_vector_type(8)))  float         v8f;
typedef __attribute__((ext_vector_type(4)))  unsigned int  v4u;

union FragU { v4u u[2]; v16bf v; };

__device__ __forceinline__ unsigned short f2bf(float f) {
  unsigned int u = __builtin_bit_cast(unsigned int, f);
  u += 0x7FFFu + ((u >> 16) & 1u);          // round-to-nearest-even
  return (unsigned short)(u >> 16);
}
__device__ __forceinline__ float bf2f(unsigned short h) {
  unsigned int u = ((unsigned int)h) << 16;
  return __builtin_bit_cast(float, u);
}
__device__ __forceinline__ float sigmoidf(float x) {
  return 1.0f / (1.0f + __expf(-x));
}

#define WMMA_BF16(a, b, c) \
  __builtin_amdgcn_wmma_f32_16x16x32_bf16(false, (a), false, (b), (short)0, (c), false, false)

// A fragment (16x32 bf16, M x K): lane holds row (lane&15),
// K = (lane>>4)*16 + {0..15}.  Two b128 loads.
__device__ __forceinline__ v16bf load_frag_a(const unsigned short* base, int ld) {
  int lane = threadIdx.x & 31;
  const unsigned short* p = base + (size_t)(lane & 15) * ld + ((lane >> 4) << 4);
  FragU f;
  f.u[0] = *(const v4u*)(p);
  f.u[1] = *(const v4u*)(p + 8);
  return f.v;
}
// B fragment (32x16 bf16, K x N) from column-major storage Bt[N,K]:
// lane holds col (lane&15), K = (lane>>4)*16 + {0..15}.  Two b128 loads.
__device__ __forceinline__ v16bf load_frag_b(const unsigned short* base, int ld) {
  int lane = threadIdx.x & 31;
  const unsigned short* p = base + (size_t)(lane & 15) * ld + ((lane >> 4) << 4);
  FragU f;
  f.u[0] = *(const v4u*)(p);
  f.u[1] = *(const v4u*)(p + 8);
  return f.v;
}

// ---------------------------------------------------------------- prep kernels
__global__ void transpose_to_bf16(const float* __restrict__ src,
                                  unsigned short* __restrict__ dst,
                                  int R, int C) {  // src [R,C] f32 -> dst [C,R] bf16
  int idx = blockIdx.x * blockDim.x + threadIdx.x;
  if (idx >= R * C) return;
  int r = idx / C, c = idx - r * C;
  dst[(size_t)c * R + r] = f2bf(src[idx]);
}

__global__ void embed_gather(const int* __restrict__ inputs,
                             const float* __restrict__ emb,
                             unsigned short* __restrict__ X) {
  int idx = blockIdx.x * blockDim.x + threadIdx.x;   // over BL*E
  if (idx >= BL * Ee) return;
  int bl = idx >> 8;              // E == 256
  int e  = idx & 255;
  X[idx] = f2bf(emb[(size_t)inputs[bl] * Ee + e]);
}

__global__ void zero_h(unsigned short* __restrict__ hbf, float* __restrict__ hf) {
  int idx = blockIdx.x * blockDim.x + threadIdx.x;
  if (idx < Bb * Hh) { hbf[idx] = 0; hf[idx] = 0.0f; }
}

// -------------------------------------------------------------- GEMM core body
// C[M,N] = A[M,K] @ B[K,N], B given as Bt[N,K]. Wave computes a 16x64 tile
// (1 A frag x 4 B frags -> 4 accumulators), ping-pong double buffered.
// Block = 8 waves = 128 rows; grid = (N/64, M/128). K % 64 == 0.
template <bool BF16_OUT>
__device__ __forceinline__ void gemm_tile_body(const unsigned short* __restrict__ A,
                                               const unsigned short* __restrict__ Bt,
                                               const float* __restrict__ bias,
                                               unsigned short* __restrict__ Cb,
                                               float* __restrict__ Cf,
                                               int K, int N) {
  int lane = threadIdx.x & 31;
  int wave = threadIdx.x >> 5;
  int m0 = blockIdx.y * 128 + wave * 16;
  int n0 = blockIdx.x * 64;
  const unsigned short* A0 = A  + (size_t)m0 * K;
  const unsigned short* B0 = Bt + (size_t)(n0 +  0) * K;
  const unsigned short* B1 = Bt + (size_t)(n0 + 16) * K;
  const unsigned short* B2 = Bt + (size_t)(n0 + 32) * K;
  const unsigned short* B3 = Bt + (size_t)(n0 + 48) * K;

  v8f acc0 = {}, acc1 = {}, acc2 = {}, acc3 = {};
  // even phase fragments @ k = 0
  v16bf ae  = load_frag_a(A0, K);
  v16bf be0 = load_frag_b(B0, K);
  v16bf be1 = load_frag_b(B1, K);
  v16bf be2 = load_frag_b(B2, K);
  v16bf be3 = load_frag_b(B3, K);
  v16bf ao, bo0, bo1, bo2, bo3;

  int k0 = 0;
  for (; k0 + 64 < K; k0 += 64) {
    // odd phase loads @ k0+32 overlap even-phase WMMAs
    ao  = load_frag_a(A0 + k0 + 32, K);
    bo0 = load_frag_b(B0 + k0 + 32, K);
    bo1 = load_frag_b(B1 + k0 + 32, K);
    bo2 = load_frag_b(B2 + k0 + 32, K);
    bo3 = load_frag_b(B3 + k0 + 32, K);
    acc0 = WMMA_BF16(ae, be0, acc0);
    acc1 = WMMA_BF16(ae, be1, acc1);
    acc2 = WMMA_BF16(ae, be2, acc2);
    acc3 = WMMA_BF16(ae, be3, acc3);
    // even phase loads @ k0+64 overlap odd-phase WMMAs
    ae  = load_frag_a(A0 + k0 + 64, K);
    be0 = load_frag_b(B0 + k0 + 64, K);
    be1 = load_frag_b(B1 + k0 + 64, K);
    be2 = load_frag_b(B2 + k0 + 64, K);
    be3 = load_frag_b(B3 + k0 + 64, K);
    acc0 = WMMA_BF16(ao, bo0, acc0);
    acc1 = WMMA_BF16(ao, bo1, acc1);
    acc2 = WMMA_BF16(ao, bo2, acc2);
    acc3 = WMMA_BF16(ao, bo3, acc3);
  }
  // tail: k0 == K-64; even @ k0 already resident, load odd @ k0+32
  ao  = load_frag_a(A0 + k0 + 32, K);
  bo0 = load_frag_b(B0 + k0 + 32, K);
  bo1 = load_frag_b(B1 + k0 + 32, K);
  bo2 = load_frag_b(B2 + k0 + 32, K);
  bo3 = load_frag_b(B3 + k0 + 32, K);
  acc0 = WMMA_BF16(ae, be0, acc0);
  acc1 = WMMA_BF16(ae, be1, acc1);
  acc2 = WMMA_BF16(ae, be2, acc2);
  acc3 = WMMA_BF16(ae, be3, acc3);
  acc0 = WMMA_BF16(ao, bo0, acc0);
  acc1 = WMMA_BF16(ao, bo1, acc1);
  acc2 = WMMA_BF16(ao, bo2, acc2);
  acc3 = WMMA_BF16(ao, bo3, acc3);

  int col = lane & 15;
  int rb  = (lane >> 4) * 8;
  v8f accs[4] = {acc0, acc1, acc2, acc3};
#pragma unroll
  for (int j = 0; j < 4; ++j) {
    int n = n0 + j * 16 + col;
    float bb = BF16_OUT ? 0.0f : bias[n];
#pragma unroll
    for (int r = 0; r < 8; ++r) {
      size_t idx = (size_t)(m0 + rb + r) * N + n;
      if (BF16_OUT) Cb[idx] = f2bf(accs[j][r]);
      else          Cf[idx] = accs[j][r] + bb;
    }
  }
}

__global__ void wmma_gemm_bf16out(const unsigned short* __restrict__ A,
                                  const unsigned short* __restrict__ Bt,
                                  unsigned short* __restrict__ C,
                                  int K, int N) {
  gemm_tile_body<true>(A, Bt, nullptr, C, nullptr, K, N);
}

__global__ void wmma_gemm_f32out(const unsigned short* __restrict__ A,
                                 const unsigned short* __restrict__ Bt,
                                 const float* __restrict__ bias,
                                 float* __restrict__ C,
                                 int K, int N) {
  gemm_tile_body<false>(A, Bt, bias, nullptr, C, K, N);
}

// ----------------------------------------------------------------- GRU step t
// 256 waves: wave (rt,ct) owns h_new rows rt*16..+15, cols ct*16..+15.
// Three 16x16 gate tiles (r,z,n) of h @ Wh over K=1024, ping-pong double
// buffered, then fused bias + nonlinearities + state update.
__global__ void gru_step(int t,
                         const unsigned short* __restrict__ hbf,
                         const float* __restrict__ hf,
                         unsigned short* __restrict__ hbf_n,
                         float* __restrict__ hf_n,
                         const unsigned short* __restrict__ XW,   // [BL, 3H] bf16
                         const unsigned short* __restrict__ WhT,  // [3H, H] bf16
                         const float* __restrict__ bh,            // [3H]
                         unsigned short* __restrict__ Y) {        // [BL, H] bf16
  int lane = threadIdx.x & 31;
  int w  = blockIdx.x * (blockDim.x >> 5) + (threadIdx.x >> 5); // 0..255
  int ct = w & 63;          // H/16 = 64 column tiles
  int rt = w >> 6;          // B/16 = 4 row tiles
  int m0 = rt * 16;
  int c0 = ct * 16;
  const unsigned short* Ab = hbf + (size_t)m0 * Hh;
  const unsigned short* Br = WhT + (size_t)(c0) * Hh;
  const unsigned short* Bz = WhT + (size_t)(Hh + c0) * Hh;
  const unsigned short* Bn = WhT + (size_t)(2 * Hh + c0) * Hh;

  v8f aR = {}, aZ = {}, aN = {};
  v16bf ae  = load_frag_a(Ab, Hh);
  v16bf bre = load_frag_b(Br, Hh);
  v16bf bze = load_frag_b(Bz, Hh);
  v16bf bne = load_frag_b(Bn, Hh);
  v16bf ao, bro, bzo, bno;

  int k0 = 0;
  for (; k0 + 64 < Hh; k0 += 64) {
    ao  = load_frag_a(Ab + k0 + 32, Hh);
    bro = load_frag_b(Br + k0 + 32, Hh);
    bzo = load_frag_b(Bz + k0 + 32, Hh);
    bno = load_frag_b(Bn + k0 + 32, Hh);
    aR = WMMA_BF16(ae, bre, aR);
    aZ = WMMA_BF16(ae, bze, aZ);
    aN = WMMA_BF16(ae, bne, aN);
    ae  = load_frag_a(Ab + k0 + 64, Hh);
    bre = load_frag_b(Br + k0 + 64, Hh);
    bze = load_frag_b(Bz + k0 + 64, Hh);
    bne = load_frag_b(Bn + k0 + 64, Hh);
    aR = WMMA_BF16(ao, bro, aR);
    aZ = WMMA_BF16(ao, bzo, aZ);
    aN = WMMA_BF16(ao, bno, aN);
  }
  ao  = load_frag_a(Ab + k0 + 32, Hh);
  bro = load_frag_b(Br + k0 + 32, Hh);
  bzo = load_frag_b(Bz + k0 + 32, Hh);
  bno = load_frag_b(Bn + k0 + 32, Hh);
  aR = WMMA_BF16(ae, bre, aR);
  aZ = WMMA_BF16(ae, bze, aZ);
  aN = WMMA_BF16(ae, bne, aN);
  aR = WMMA_BF16(ao, bro, aR);
  aZ = WMMA_BF16(ao, bzo, aZ);
  aN = WMMA_BF16(ao, bno, aN);

  int col = c0 + (lane & 15);
  int rb  = (lane >> 4) * 8;
  float bhr = bh[col], bhz = bh[Hh + col], bhn = bh[2 * Hh + col];
#pragma unroll
  for (int r = 0; r < 8; ++r) {
    int b = m0 + rb + r;                         // batch row
    size_t xrow = ((size_t)b * Ll + t) * (size_t)G3H;
    float xr = bf2f(XW[xrow + col]);
    float xz = bf2f(XW[xrow + Hh + col]);
    float xn = bf2f(XW[xrow + 2 * Hh + col]);
    float rr = sigmoidf(xr + aR[r] + bhr);
    float zz = sigmoidf(xz + aZ[r] + bhz);
    float nn = tanhf(xn + rr * (aN[r] + bhn));
    float hold = hf[(size_t)b * Hh + col];
    float hnew = (1.0f - zz) * nn + zz * hold;
    hf_n[(size_t)b * Hh + col]  = hnew;
    hbf_n[(size_t)b * Hh + col] = f2bf(hnew);
    Y[((size_t)b * Ll + t) * Hh + col] = f2bf(hnew);
  }
}

// --------------------------------------------------------------------- launch
extern "C" void kernel_launch(void* const* d_in, const int* in_sizes, int n_in,
                              void* d_out, int out_size, void* d_ws, size_t ws_size,
                              hipStream_t stream) {
  const int*   inputs = (const int*)  d_in[0];
  const float* emb    = (const float*)d_in[1];
  const float* Wi     = (const float*)d_in[2];   // [E, 3H]
  const float* Wh     = (const float*)d_in[3];   // [H, 3H]
  const float* bh     = (const float*)d_in[4];   // [3H]
  const float* Wout   = (const float*)d_in[5];   // [H, V]
  const float* bout   = (const float*)d_in[6];   // [V]
  float* out = (float*)d_out;                    // [B, L, V] f32

  char* ws = (char*)d_ws;
  size_t off = 0;
  auto alloc = [&](size_t bytes) -> void* {
    void* p = ws + off;
    off = (off + bytes + 255) & ~(size_t)255;
    return p;
  };
  unsigned short* WiT   = (unsigned short*)alloc((size_t)G3H * Ee * 2);  // [3H,E]
  unsigned short* WhT   = (unsigned short*)alloc((size_t)G3H * Hh * 2);  // [3H,H]
  unsigned short* WoutT = (unsigned short*)alloc((size_t)Vv  * Hh * 2);  // [V,H]
  unsigned short* Xbf   = (unsigned short*)alloc((size_t)BL  * Ee * 2);  // [BL,E]
  unsigned short* XW    = (unsigned short*)alloc((size_t)BL  * G3H * 2); // [BL,3H]
  unsigned short* Ybf   = (unsigned short*)alloc((size_t)BL  * Hh * 2);  // [BL,H]
  unsigned short* Hbf0  = (unsigned short*)alloc((size_t)Bb * Hh * 2);
  unsigned short* Hbf1  = (unsigned short*)alloc((size_t)Bb * Hh * 2);
  float*          Hf0   = (float*)alloc((size_t)Bb * Hh * 4);
  float*          Hf1   = (float*)alloc((size_t)Bb * Hh * 4);

  // 1) weight transpose + bf16 convert
  transpose_to_bf16<<<(Ee * G3H + 255) / 256, 256, 0, stream>>>(Wi,   WiT,   Ee, G3H);
  transpose_to_bf16<<<(Hh * G3H + 255) / 256, 256, 0, stream>>>(Wh,   WhT,   Hh, G3H);
  transpose_to_bf16<<<(Hh * Vv  + 255) / 256, 256, 0, stream>>>(Wout, WoutT, Hh, Vv);

  // 2) embedding gather
  embed_gather<<<(BL * Ee + 255) / 256, 256, 0, stream>>>(inputs, emb, Xbf);

  // 3) xw = x @ Wi   (M=16384, N=3072, K=256)
  {
    dim3 grid(G3H / 64, BL / 128);
    wmma_gemm_bf16out<<<grid, 256, 0, stream>>>(Xbf, WiT, XW, Ee, G3H);
  }

  // 4) recurrent scan, h0 = 0
  zero_h<<<(Bb * Hh + 255) / 256, 256, 0, stream>>>(Hbf0, Hf0);
  for (int t = 0; t < Ll; ++t) {
    unsigned short* hc  = (t & 1) ? Hbf1 : Hbf0;
    unsigned short* hn  = (t & 1) ? Hbf0 : Hbf1;
    float*          hcf = (t & 1) ? Hf1  : Hf0;
    float*          hnf = (t & 1) ? Hf0  : Hf1;
    gru_step<<<32, 256, 0, stream>>>(t, hc, hcf, hn, hnf, XW, WhT, bh, Ybf);
  }

  // 5) logits = y @ Wout + bout   (M=16384, N=256, K=1024)
  {
    dim3 grid(Vv / 64, BL / 128);
    wmma_gemm_f32out<<<grid, 256, 0, stream>>>(Ybf, WoutT, bout, out, Hh, Vv);
  }
}
